// ResidualMamba_17437567222400
// MI455X (gfx1250) — matrix operand order repaired and checked by reading, not compiled, not run on hardware
//
#include <hip/hip_runtime.h>
#include <hip/hip_bf16.h>

// ---------------------------------------------------------------------------
// Problem constants (from reference)
// ---------------------------------------------------------------------------
#define B_ 2
#define L_ 2048
#define DM_ 1024
#define E_ 2048
#define N_ 16
#define K_ 4
#define R_ 64
#define M_ (B_ * L_)          // 4096 rows
#define EPS_ 1e-5f

// Use the Tensor Data Mover for GEMM A-tile staging when the builtin exists.
#define CDNA5_USE_TDM 1

typedef __attribute__((ext_vector_type(16))) __bf16 v16bf;
typedef __attribute__((ext_vector_type(8)))  float  v8f;
typedef __attribute__((ext_vector_type(4)))  unsigned int v4u;
typedef __attribute__((ext_vector_type(8)))  int v8i;
typedef __attribute__((ext_vector_type(4)))  int v4i;

union FragBF16 {
    uint4 q[2];
    v16bf v;
};

__device__ __forceinline__ unsigned short f32_to_bf16(float f) {
    union { float f; unsigned int u; } x;
    x.f = f;
    unsigned int r = x.u + 0x7FFFu + ((x.u >> 16) & 1u);  // round-to-nearest-even
    return (unsigned short)(r >> 16);
}

__device__ __forceinline__ float silu_f(float v) {
    return v / (1.0f + __expf(-v));
}

// ---------------------------------------------------------------------------
// 0) f32 -> bf16 copy (weight prep; runs once per launch, off critical path)
// ---------------------------------------------------------------------------
__global__ __launch_bounds__(256) void cvt_f32_bf16_kernel(
    const float* __restrict__ src, unsigned short* __restrict__ dst)
{
    const size_t idx = (size_t)blockIdx.x * 256 + threadIdx.x;
    dst[idx] = f32_to_bf16(src[idx]);
}

// ---------------------------------------------------------------------------
// 1) RMSNorm: xn_bf16[r, c] = x[r,c] * rsqrt(mean(x^2)+eps) * norm_w[c]
// ---------------------------------------------------------------------------
__global__ __launch_bounds__(256) void rmsnorm_kernel(
    const float* __restrict__ x, const float* __restrict__ norm_w,
    unsigned short* __restrict__ xn_bf16)
{
    __shared__ float red[256];
    const int r = blockIdx.x;
    const int tid = threadIdx.x;
    float vals[4];
    float s = 0.f;
#pragma unroll
    for (int i = 0; i < 4; ++i) {
        const int c = tid + i * 256;
        const float v = x[(size_t)r * DM_ + c];
        vals[i] = v;
        s += v * v;
    }
    red[tid] = s;
    __syncthreads();
    for (int off = 128; off > 0; off >>= 1) {
        if (tid < off) red[tid] += red[tid + off];
        __syncthreads();
    }
    const float scale = rsqrtf(red[0] * (1.0f / (float)DM_) + EPS_);
#pragma unroll
    for (int i = 0; i < 4; ++i) {
        const int c = tid + i * 256;
        xn_bf16[(size_t)r * DM_ + c] = f32_to_bf16(vals[i] * scale * norm_w[c]);
    }
}

// ---------------------------------------------------------------------------
// 2) WMMA bf16 GEMM: out[M, NO] = A[M,KD] * Wb[NO,KD]^T  (+ epilogue)
//    Workgroup tile 128x64, 8 waves as 4x2, each wave register-tiles 2x2
//    16x16 subtiles -> 4 WMMAs per LDS stage with fragment reuse.
//    A tile staged by the Tensor Data Mover (TDM) when available.
//    EP: 0 = plain f32 store, 1 = +bias then softplus, 2 = +residual
//    GUARD: bounds-check columns against NO (only needed for x_proj, NO=96)
// ---------------------------------------------------------------------------
#define TM 128
#define TN 64
#define LP 40   // LDS row pitch in bf16 elements (80 B; keeps 16B alignment)

template <int EP, bool GUARD>
__global__ __launch_bounds__(256) void gemm_bf16_wmma(
    const unsigned short* __restrict__ A,   // [M, KD] bf16
    const unsigned short* __restrict__ Wb,  // [NO, KD] bf16
    float* __restrict__ out,                // [M, ldo] f32
    const float* __restrict__ bias,         // EP==1
    const float* __restrict__ resid,        // EP==2
    int KD, int NO, int ldo)
{
    __shared__ unsigned short lsA[TM * LP];  // 10240 B
    __shared__ unsigned short lsB[TN * LP];  //  5120 B

    const int tid  = threadIdx.x;
    const int lane = tid & 31;
    const int wave = tid >> 5;
    const int wr   = wave & 3;   // wave row  -> m offset 32*wr
    const int wc   = wave >> 2;  // wave col  -> n offset 32*wc
    const int tm   = blockIdx.x * TM;
    const int tn   = blockIdx.y * TN;

    // staging assignments
    const int b_n = tid >> 2;          // 0..63
    const int b_c = (tid & 3) << 3;    // 0,8,16,24 (8 bf16 = 16B)
    const bool b_ok = GUARD ? ((tn + b_n) < NO) : true;

#if CDNA5_USE_TDM && __has_builtin(__builtin_amdgcn_tensor_load_to_lds)
    // TDM D# group1 (constant over K loop): data_size=2B, pad 16B after each
    // 64B row chunk (pad_interval=3 -> 8x8B=64B, pad_amount=3 -> 4 DWORDs)
    // => LDS pitch 80B == LP. Tile = 32 (dim0) x 128 rows (dim1), stride KD.
    const unsigned int lds_base = (unsigned int)(unsigned long long)(void*)lsA;
    v8i tdm_g1;
    tdm_g1[0] = (int)((1u << 16) | (1u << 20) | (3u << 22) | (3u << 25));
    tdm_g1[1] = (int)(((unsigned int)KD & 0xFFFFu) << 16);    // tensor_dim0[15:0]
    tdm_g1[2] = (int)(((unsigned int)M_ & 0xFFFFu) << 16);    // tensor_dim1[15:0]
    tdm_g1[3] = (int)(((unsigned int)M_ >> 16) | (32u << 16)); // tile_dim0 = 32
    tdm_g1[4] = (int)TM;                                       // tile_dim1 = 128
    tdm_g1[5] = (int)KD;                                       // dim0 stride
    tdm_g1[6] = 0;
    tdm_g1[7] = 0;
#else
    const int a_r = tid >> 1;          // 0..127
    const int a_c = (tid & 1) << 4;    // 0 or 16 (16 bf16 = 32B)
#endif

    v8f acc[2][2] = {{{}, {}}, {{}, {}}};
    const int ksteps = KD >> 5;  // K chunk = 32

    const int half = lane >> 4;
    const int lmod = lane & 15;

    for (int ks = 0; ks < ksteps; ++ks) {
        const int k0 = ks << 5;

        // --- stage B tile: global -> regs ---
        uint4 bv = {0u, 0u, 0u, 0u};
        if (b_ok) {
            bv = *(const uint4*)(Wb + (size_t)(tn + b_n) * KD + k0 + b_c);
            if (ks + 1 < ksteps)
                __builtin_prefetch(Wb + (size_t)(tn + b_n) * KD + k0 + 32 + b_c, 0, 1);
        }

#if CDNA5_USE_TDM && __has_builtin(__builtin_amdgcn_tensor_load_to_lds)
        __syncthreads();  // previous iteration's LDS reads complete
        if (wave == 0) {  // wave-uniform: one TDM issue per workgroup
            const unsigned long long ga =
                (unsigned long long)(const void*)(A + (size_t)tm * KD + k0);
            v4u g0;
            g0[0] = 1u;                    // count=1, user mode
            g0[1] = lds_base;              // lds_addr (bytes)
            g0[2] = (unsigned int)ga;      // global_addr[31:0]
            g0[3] = (unsigned int)((ga >> 32) & 0x1FFFFFFull) | (2u << 30); // type=2
            v4i z4 = {0, 0, 0, 0};
#if __clang_major__ >= 23
            v8i z8 = {0, 0, 0, 0, 0, 0, 0, 0};
            __builtin_amdgcn_tensor_load_to_lds(g0, tdm_g1, z4, z4, z8, 0);
#else
            __builtin_amdgcn_tensor_load_to_lds(g0, tdm_g1, z4, z4, 0);
#endif
            __builtin_amdgcn_s_wait_tensorcnt(0);
        }
        *(uint4*)(lsB + b_n * LP + b_c) = bv;
        __syncthreads();  // publish TDM A-tile + lsB to all waves
#else
        const uint4* gA = (const uint4*)(A + (size_t)(tm + a_r) * KD + k0 + a_c);
        const uint4 av0 = gA[0];
        const uint4 av1 = gA[1];
        if (ks + 1 < ksteps)
            __builtin_prefetch(A + (size_t)(tm + a_r) * KD + k0 + 32 + a_c, 0, 1);
        __syncthreads();
        {
            uint4* sA = (uint4*)(lsA + a_r * LP + a_c);
            sA[0] = av0;
            sA[1] = av1;
            *(uint4*)(lsB + b_n * LP + b_c) = bv;
        }
        __syncthreads();
#endif

        // --- fragments (documented gfx1250 layouts) ---
        FragBF16 fa[2], fb[2];
#pragma unroll
        for (int i = 0; i < 2; ++i) {
            const int mrow = (wr << 5) + (i << 4) + lmod;
            // A 16x32: elem j<8 -> K = half*8 + j ; j>=8 -> K = 16 + half*8 + (j-8)
            fa[i].q[0] = *(const uint4*)(lsA + mrow * LP + (half << 3));
            fa[i].q[1] = *(const uint4*)(lsA + mrow * LP + 16 + (half << 3));
        }
#pragma unroll
        for (int j = 0; j < 2; ++j) {
            const int ncol = (wc << 5) + (j << 4) + lmod;
            // B 32x16: column n = lane%16, elem i<8 -> K = half*8 + i ; i>=8 -> K = 16 + half*8 + (i-8)
            fb[j].q[0] = *(const uint4*)(lsB + ncol * LP + (half << 3));
            fb[j].q[1] = *(const uint4*)(lsB + ncol * LP + 16 + (half << 3));
        }

#pragma unroll
        for (int i = 0; i < 2; ++i)
#pragma unroll
            for (int j = 0; j < 2; ++j)
                acc[i][j] = __builtin_amdgcn_wmma_f32_16x16x32_bf16(
                    false, fa[i].v, false, fb[j].v,
                    (short)0, acc[i][j], false, false);
    }

    // --- epilogue: C/D lane -> (row = r + 8*half, col = lane%16) ---
#pragma unroll
    for (int j = 0; j < 2; ++j) {
        const int col = tn + (wc << 5) + (j << 4) + lmod;
        if (GUARD && col >= NO) continue;
        const float bcol = (EP == 1) ? bias[col] : 0.f;
#pragma unroll
        for (int i = 0; i < 2; ++i) {
#pragma unroll
            for (int r = 0; r < 8; ++r) {
                const int row = tm + (wr << 5) + (i << 4) + r + (half << 3);
                float v = acc[i][j][r];
                if (EP == 1) {
                    v += bcol;
                    v = (v > 20.f) ? v : log1pf(__expf(v));  // softplus
                }
                if (EP == 2) {
                    v += resid[(size_t)row * ldo + col];
                }
                out[(size_t)row * ldo + col] = v;
            }
        }
    }
}

// ---------------------------------------------------------------------------
// 3) Causal depthwise conv (K=4) + bias + SiLU.
// ---------------------------------------------------------------------------
__global__ __launch_bounds__(256) void conv_silu_kernel(
    const float* __restrict__ xz, const float* __restrict__ conv_w,
    const float* __restrict__ conv_b,
    float* __restrict__ u_f32, unsigned short* __restrict__ u_bf16)
{
    const size_t idx = (size_t)blockIdx.x * 256 + threadIdx.x;  // over M*E
    const int e = (int)(idx % E_);
    const int m = (int)(idx / E_);
    const int l = m % L_;
    float acc = conv_b[e];
#pragma unroll
    for (int j = 0; j < K_; ++j) {
        const int li = l - (K_ - 1) + j;
        if (li >= 0) acc += xz[(size_t)(m - (K_ - 1 - j)) * (2 * E_) + e] * conv_w[e * K_ + j];
    }
    const float v = silu_f(acc);
    u_f32[idx]  = v;
    u_bf16[idx] = f32_to_bf16(v);
}

// ---------------------------------------------------------------------------
// 4) Convert dt_low slice (x_dbl[:, 0:R]) to bf16.
// ---------------------------------------------------------------------------
__global__ __launch_bounds__(256) void cvt_dtlow_kernel(
    const float* __restrict__ x_dbl, unsigned short* __restrict__ dtlow)
{
    const size_t idx = (size_t)blockIdx.x * 256 + threadIdx.x;  // over M*R
    const int m = (int)(idx / R_);
    const int j = (int)(idx % R_);
    dtlow[idx] = f32_to_bf16(x_dbl[(size_t)m * (R_ + 2 * N_) + j]);
}

// ---------------------------------------------------------------------------
// 5) Selective scan. One thread per (b, e) channel; h[N], A[N] in registers.
// ---------------------------------------------------------------------------
__global__ __launch_bounds__(256) void scan_kernel(
    const float* __restrict__ u_f32,   // [M, E]
    const float* __restrict__ dtb,     // [M, E]
    const float* __restrict__ x_dbl,   // [M, R+2N]
    const float* __restrict__ A_log,   // [E, N]
    float* __restrict__ ys)            // [M, E]
{
    __shared__ float shBC[2 * N_];
    const int tid = threadIdx.x;
    const int b   = blockIdx.x >> 3;          // E/256 = 8 blocks per batch
    const int eb  = blockIdx.x & 7;
    const int e   = eb * 256 + tid;

    float A[N_], h[N_];
#pragma unroll
    for (int n = 0; n < N_; ++n) {
        A[n] = -__expf(A_log[(size_t)e * N_ + n]);
        h[n] = 0.f;
    }

    for (int l = 0; l < L_; ++l) {
        const size_t row = (size_t)b * L_ + l;
        if (tid < 2 * N_) shBC[tid] = x_dbl[row * (R_ + 2 * N_) + R_ + tid];
        __syncthreads();

        const float dt = dtb[row * E_ + e];
        const float du = dt * u_f32[row * E_ + e];
        float y = 0.f;
#pragma unroll
        for (int n = 0; n < N_; ++n) {
            const float dA = __expf(dt * A[n]);
            h[n] = h[n] * dA + du * shBC[n];
            y += h[n] * shBC[N_ + n];
        }
        ys[row * E_ + e] = y;
        __syncthreads();
    }
}

// ---------------------------------------------------------------------------
// 6) y = (ys + u*D) * silu(z)  -> bf16  (z = xz[:, E:2E])
// ---------------------------------------------------------------------------
__global__ __launch_bounds__(256) void ymul_kernel(
    const float* __restrict__ ys, const float* __restrict__ u_f32,
    const float* __restrict__ xz, const float* __restrict__ D,
    unsigned short* __restrict__ y_bf16)
{
    const size_t idx = (size_t)blockIdx.x * 256 + threadIdx.x;  // over M*E
    const int e = (int)(idx % E_);
    const int m = (int)(idx / E_);
    const float z = xz[(size_t)m * (2 * E_) + E_ + e];
    const float y = (ys[idx] + u_f32[idx] * D[e]) * silu_f(z);
    y_bf16[idx] = f32_to_bf16(y);
}

// ---------------------------------------------------------------------------
// Launch
// ---------------------------------------------------------------------------
extern "C" void kernel_launch(void* const* d_in, const int* in_sizes, int n_in,
                              void* d_out, int out_size, void* d_ws, size_t ws_size,
                              hipStream_t stream) {
    (void)in_sizes; (void)n_in; (void)out_size; (void)ws_size;

    const float* x         = (const float*)d_in[0];   // [B,L,DM]
    const float* norm_w    = (const float*)d_in[1];   // [DM]
    const float* in_proj_w = (const float*)d_in[2];   // [2E, DM]
    const float* conv_w    = (const float*)d_in[3];   // [E, K]
    const float* conv_b    = (const float*)d_in[4];   // [E]
    const float* x_proj_w  = (const float*)d_in[5];   // [R+2N, E]
    const float* dt_proj_w = (const float*)d_in[6];   // [E, R]
    const float* dt_proj_b = (const float*)d_in[7];   // [E]
    const float* A_log     = (const float*)d_in[8];   // [E, N]
    const float* Dv        = (const float*)d_in[9];   // [E]
    const float* out_proj_w= (const float*)d_in[10];  // [DM, E]
    float* out = (float*)d_out;                       // [B,L,DM]

    // ---- workspace layout ----
    char* ws = (char*)d_ws;
    size_t off = 0;
    unsigned short* xn_bf16 = (unsigned short*)(ws + off); off += (size_t)M_ * DM_ * 2;        // 8 MB
    float*          xz      = (float*)(ws + off);          off += (size_t)M_ * 2 * E_ * 4;     // 64 MB
    float*          u_f32   = (float*)(ws + off);          off += (size_t)M_ * E_ * 4;         // 32 MB
    unsigned short* u_bf16  = (unsigned short*)(ws + off); off += (size_t)M_ * E_ * 2;         // 16 MB
    float*          x_dbl   = (float*)(ws + off);          off += (size_t)M_ * (R_ + 2*N_) * 4;// 1.5 MB
    off = (off + 255) & ~(size_t)255;
    unsigned short* dtlow   = (unsigned short*)(ws + off); off += (size_t)M_ * R_ * 2;         // 0.5 MB
    off = (off + 255) & ~(size_t)255;
    float*          dtb     = (float*)(ws + off);          off += (size_t)M_ * E_ * 4;         // 32 MB
    float*          ysb     = (float*)(ws + off);          off += (size_t)M_ * E_ * 4;         // 32 MB
    unsigned short* y_bf16  = (unsigned short*)(ws + off); off += (size_t)M_ * E_ * 2;         // 16 MB
    // bf16 weight copies
    unsigned short* w_in  = (unsigned short*)(ws + off); off += (size_t)(2*E_) * DM_ * 2;      // 8 MB
    unsigned short* w_x   = (unsigned short*)(ws + off); off += (size_t)(R_ + 2*N_) * E_ * 2;  // 0.375 MB
    unsigned short* w_dt  = (unsigned short*)(ws + off); off += (size_t)E_ * R_ * 2;           // 0.25 MB
    unsigned short* w_out = (unsigned short*)(ws + off); off += (size_t)DM_ * E_ * 2;          // 4 MB

    // 0) weight prep: f32 -> bf16 (all sizes are multiples of 256)
    cvt_f32_bf16_kernel<<<(2 * E_ * DM_) / 256, 256, 0, stream>>>(in_proj_w, w_in);
    cvt_f32_bf16_kernel<<<((R_ + 2 * N_) * E_) / 256, 256, 0, stream>>>(x_proj_w, w_x);
    cvt_f32_bf16_kernel<<<(E_ * R_) / 256, 256, 0, stream>>>(dt_proj_w, w_dt);
    cvt_f32_bf16_kernel<<<(DM_ * E_) / 256, 256, 0, stream>>>(out_proj_w, w_out);

    // 1) RMSNorm -> bf16
    rmsnorm_kernel<<<M_, 256, 0, stream>>>(x, norm_w, xn_bf16);

    // 2) in_proj: [M,DM] x [2E,DM]^T -> xz [M,2E]
    gemm_bf16_wmma<0, false><<<dim3(M_ / TM, (2 * E_) / TN), 256, 0, stream>>>(
        xn_bf16, w_in, xz, nullptr, nullptr, DM_, 2 * E_, 2 * E_);

    // 3) depthwise conv + SiLU -> u (f32 + bf16)
    conv_silu_kernel<<<(M_ * E_) / 256, 256, 0, stream>>>(xz, conv_w, conv_b, u_f32, u_bf16);

    // 4) x_proj: [M,E] x [R+2N,E]^T -> x_dbl [M,96]   (NO=96 -> guarded)
    gemm_bf16_wmma<0, true><<<dim3(M_ / TM, (R_ + 2 * N_ + TN - 1) / TN), 256, 0, stream>>>(
        u_bf16, w_x, x_dbl, nullptr, nullptr, E_, R_ + 2 * N_, R_ + 2 * N_);

    // 5) dt = softplus(dt_low x dt_proj_w^T + b) -> [M,E]
    cvt_dtlow_kernel<<<(M_ * R_) / 256, 256, 0, stream>>>(x_dbl, dtlow);
    gemm_bf16_wmma<1, false><<<dim3(M_ / TM, E_ / TN), 256, 0, stream>>>(
        dtlow, w_dt, dtb, dt_proj_b, nullptr, R_, E_, E_);

    // 6) selective scan
    scan_kernel<<<B_ * (E_ / 256), 256, 0, stream>>>(u_f32, dtb, x_dbl, A_log, ysb);

    // 7) y = (ys + u*D) * silu(z) -> bf16
    ymul_kernel<<<(M_ * E_) / 256, 256, 0, stream>>>(ysb, u_f32, xz, Dv, y_bf16);

    // 8) out_proj + residual -> d_out
    gemm_bf16_wmma<2, false><<<dim3(M_ / TM, DM_ / TN), 256, 0, stream>>>(
        y_bf16, w_out, out, nullptr, x, E_, DM_, DM_);
}
